// ElementTypeLinear_2765958939069
// MI455X (gfx1250) — matrix operand order repaired and checked by reading, not compiled, not run on hardware
//
#include <hip/hip_runtime.h>
#include <hip/hip_bf16.h>
#include <stdint.h>

// CDNA5 / gfx1250: wave32, WMMA 16x16x32 bf16 -> f32
typedef __attribute__((ext_vector_type(16))) __bf16 v16bf;
typedef __attribute__((ext_vector_type(8)))  float  v8f;

#define NTYPES 5
#define D      256   // D_IN == D_OUT
#define BM     64    // atoms per GEMM block
#define KTILES 8     // 256 / 32
#define NTILES 16    // 256 / 16
// per-(type,kchunk) packed-B chunk: 2(hi/lo) * 16 ntiles * 32 lanes * 16 bf16 = 16384 u16 = 32KB
#define BCHUNK_U16 (2 * NTILES * 32 * 16)

__device__ __forceinline__ unsigned short f2bf_rne(float f) {
    unsigned u = __float_as_uint(f);
    u += 0x7FFFu + ((u >> 16) & 1u);      // round-to-nearest-even (finite inputs)
    return (unsigned short)(u >> 16);
}
__device__ __forceinline__ float bf2f(unsigned short b) {
    return __uint_as_float(((unsigned)b) << 16);
}

union BF16Frag {
    v16bf v;
    __bf16 h[16];
    uint4 q[2];
};

// ---------------- sort-by-type machinery ----------------
// hdr[0..4]=counts, hdr[8..12]=offsets, hdr[16..20]=cursors
__global__ void etl_zero(int* hdr) {
    if (threadIdx.x < 32) hdr[threadIdx.x] = 0;
}

__global__ void etl_hist(const int* __restrict__ z, const int* __restrict__ table,
                         int* hdr, int n) {
    __shared__ int lc[NTYPES];
    const int tid = threadIdx.x;
    if (tid < NTYPES) lc[tid] = 0;
    __syncthreads();
    const int i = blockIdx.x * blockDim.x + tid;
    if (i < n) atomicAdd(&lc[table[z[i]]], 1);
    __syncthreads();
    if (tid < NTYPES) atomicAdd(&hdr[tid], lc[tid]);
}

__global__ void etl_scan(int* hdr) {
    if (threadIdx.x == 0 && blockIdx.x == 0) {
        int s = 0;
        for (int t = 0; t < NTYPES; ++t) { hdr[8 + t] = s; hdr[16 + t] = s; s += hdr[t]; }
    }
}

__global__ void etl_scatter(const int* __restrict__ z, const int* __restrict__ table,
                            int* hdr, int* __restrict__ perm, int n) {
    __shared__ int lc[NTYPES], lbase[NTYPES], lcur[NTYPES];
    const int tid = threadIdx.x;
    if (tid < NTYPES) { lc[tid] = 0; lcur[tid] = 0; }
    __syncthreads();
    const int i = blockIdx.x * blockDim.x + tid;
    int t = 0;
    if (i < n) { t = table[z[i]]; atomicAdd(&lc[t], 1); }
    __syncthreads();
    if (tid < NTYPES) lbase[tid] = atomicAdd(&hdr[16 + tid], lc[tid]);
    __syncthreads();
    if (i < n) {
        int r = atomicAdd(&lcur[t], 1);
        perm[lbase[t] + r] = i;
    }
}

// ---------------- weight pre-split into B-fragment layout ----------------
// wsB layout: [t][kc][hl][nt][lane][e]
// B 32x16 bf16 layout (ISA): lanes 0-15 -> col=lane,    K = kc*32 + (e<8 ? e : e+8)
//                            lanes 16-31 -> col=lane-16, K = kc*32 + 8 + (e<8 ? e : e+8)
__global__ void etl_pack(const float* __restrict__ w, unsigned short* __restrict__ wsB) {
    const int id   = blockIdx.x * blockDim.x + threadIdx.x;
    const int e    = id & 15;
    const int lane = (id >> 4) & 31;
    const int nt   = (id >> 9) & 15;
    const int hl   = (id >> 13) & 1;
    const int kc   = (id >> 14) & 7;
    const int t    = id >> 17;
    if (t >= NTYPES) return;
    const int ncol = nt * 16 + (lane & 15);
    const int k    = kc * 32 + ((lane < 16) ? 0 : 8) + ((e < 8) ? e : e + 8);
    const float wf = w[((size_t)t * D + k) * D + ncol];
    const unsigned short hb = f2bf_rne(wf);
    wsB[id] = hl ? f2bf_rne(wf - bf2f(hb)) : hb;
}

// ---------------- grouped GEMM via bf16 WMMA (hi/lo split, f32 accumulate) ----------------
__global__ __launch_bounds__(256) void etl_gemm(
    const float* __restrict__ x, const int* __restrict__ perm,
    const int* __restrict__ hdr, const unsigned short* __restrict__ wsB,
    const float* __restrict__ bias, float* __restrict__ out)
{
    const int t        = blockIdx.y;
    const int segCnt   = hdr[t];
    const int segBase  = hdr[8 + t];
    const int rowStart = blockIdx.x * BM;
    if (rowStart >= segCnt) return;           // block-uniform: EXEC stays all-ones below

    __shared__ unsigned short ldsB[2][BCHUNK_U16];   // 2 x 32KB (double buffer)

    const int tid  = threadIdx.x;
    const int lane = tid & 31;
    const int wave = tid >> 5;
    const int rt   = wave >> 1;   // row tile 0..3 (16 atoms each)
    const int ch   = wave & 1;    // column half (8 ntiles each)
    const int li   = lane & 15;

    // A-matrix row for this lane (16-bit A layout: both lane halves hold rows 0..15)
    const int apos = rowStart + rt * 16 + li;
    const int arow = perm[segBase + ((apos < segCnt) ? apos : 0)];
    const float* xrow = x + (size_t)arow * D;
    // lanes 0-15: K in [0:8)∪[16:24); lanes 16-31: K in [8:16)∪[24:32) per 32-chunk
    const int kshiftA = (lane < 16) ? 0 : 8;

    v8f acc[8] = {};

    const uint4* wsT = (const uint4*)(wsB + (size_t)t * (KTILES * BCHUNK_U16));

    // prologue: stage k-chunk 0 into buffer 0
    {
        const uint4* src = wsT;
        uint4* dst = (uint4*)ldsB[0];
#pragma unroll
        for (int i = 0; i < 8; ++i)
            dst[tid + i * 256] = src[tid + i * 256];
    }
    __syncthreads();

    for (int kc = 0; kc < KTILES; ++kc) {
        const unsigned short* buf = ldsB[kc & 1];

        // issue next chunk's global loads early (latency hidden behind WMMAs)
        uint4 stage[8];
        if (kc + 1 < KTILES) {
            const uint4* src = wsT + (size_t)(kc + 1) * (BCHUNK_U16 / 8);
#pragma unroll
            for (int i = 0; i < 8; ++i)
                stage[i] = src[tid + i * 256];
        }

        // load 16 fp32 of this row's K-chunk and split into hi/lo bf16 (native cvt)
        const int kab = kc * 32 + kshiftA;
        const float4 f0 = *(const float4*)(xrow + kab);
        const float4 f1 = *(const float4*)(xrow + kab + 4);
        const float4 f2 = *(const float4*)(xrow + kab + 16);
        const float4 f3 = *(const float4*)(xrow + kab + 20);
        const float fs[16] = { f0.x,f0.y,f0.z,f0.w, f1.x,f1.y,f1.z,f1.w,
                               f2.x,f2.y,f2.z,f2.w, f3.x,f3.y,f3.z,f3.w };
        BF16Frag ah, al;
#pragma unroll
        for (int e = 0; e < 16; ++e) {
            const __bf16 hb = (__bf16)fs[e];
            ah.h[e] = hb;
            al.h[e] = (__bf16)(fs[e] - (float)hb);
        }

#pragma unroll
        for (int j = 0; j < 8; ++j) {
            const int nt = ch * 8 + j;
            BF16Frag bh, bl;
            const int fbH = ((0 * NTILES + nt) * 32 + lane) * 16;
            const int fbL = ((1 * NTILES + nt) * 32 + lane) * 16;
            bh.q[0] = *(const uint4*)&buf[fbH];
            bh.q[1] = *(const uint4*)&buf[fbH + 8];
            bl.q[0] = *(const uint4*)&buf[fbL];
            bl.q[1] = *(const uint4*)&buf[fbL + 8];
            // x_hi*W_hi + x_lo*W_hi + x_hi*W_lo  (f32 accumulate)
            acc[j] = __builtin_amdgcn_wmma_f32_16x16x32_bf16(false, ah.v, false, bh.v,
                                                             (short)0, acc[j], false, false);
            acc[j] = __builtin_amdgcn_wmma_f32_16x16x32_bf16(false, al.v, false, bh.v,
                                                             (short)0, acc[j], false, false);
            acc[j] = __builtin_amdgcn_wmma_f32_16x16x32_bf16(false, ah.v, false, bl.v,
                                                             (short)0, acc[j], false, false);
        }

        // write next chunk into the other buffer (no conflict with current reads)
        if (kc + 1 < KTILES) {
            uint4* dst = (uint4*)ldsB[(kc + 1) & 1];
#pragma unroll
            for (int i = 0; i < 8; ++i)
                dst[tid + i * 256] = stage[i];
        }
        __syncthreads();
    }

    // C/D layout: VGPR i -> M = i + 8*(lane>=16), N = lane&15 within tile
    const int mAdd = (lane >> 4) * 8;
    int grow[8];
#pragma unroll
    for (int i = 0; i < 8; ++i) {
        const int pm = rowStart + rt * 16 + mAdd + i;
        grow[i] = (pm < segCnt) ? perm[segBase + pm] : -1;
    }
#pragma unroll
    for (int j = 0; j < 8; ++j) {
        const int col = (ch * 8 + j) * 16 + li;
        const float bv = bias[t * D + col];
#pragma unroll
        for (int i = 0; i < 8; ++i) {
            if (grow[i] >= 0)
                out[(size_t)grow[i] * D + col] = acc[j][i] + bv;
        }
    }
}

// ---------------- launch ----------------
extern "C" void kernel_launch(void* const* d_in, const int* in_sizes, int n_in,
                              void* d_out, int out_size, void* d_ws, size_t ws_size,
                              hipStream_t stream) {
    const float* x     = (const float*)d_in[0];
    const int*   z     = (const int*)d_in[1];
    const float* w     = (const float*)d_in[2];
    const float* bias  = (const float*)d_in[3];
    const int*   table = (const int*)d_in[4];
    const int    n     = in_sizes[1];          // N_ATOMS
    float* out = (float*)d_out;

    int* hdr  = (int*)d_ws;
    int* perm = hdr + 32;
    uintptr_t p = (uintptr_t)(perm + n);
    p = (p + 15) & ~(uintptr_t)15;
    unsigned short* wsB = (unsigned short*)p;

    etl_zero<<<1, 32, 0, stream>>>(hdr);
    etl_hist<<<(n + 255) / 256, 256, 0, stream>>>(z, table, hdr, n);
    etl_scan<<<1, 1, 0, stream>>>(hdr);
    etl_scatter<<<(n + 255) / 256, 256, 0, stream>>>(z, table, hdr, perm, n);
    etl_pack<<<(NTYPES << 17) / 256, 256, 0, stream>>>(w, wsB);

    dim3 grid((n + BM - 1) / BM, NTYPES);
    etl_gemm<<<grid, 256, 0, stream>>>(x, perm, hdr, wsB, bias, out);
}
